// ResidualAttentionBlock_prefix_15023795601989
// MI455X (gfx1250) — matrix operand order repaired and checked by reading, not compiled, not run on hardware
//
#include <hip/hip_runtime.h>

// ---------------------------------------------------------------------------
// CDNA5 (gfx1250, wave32) WMMA bf16 implementation of the residual attention
// block.  All matmuls go through v_wmma_f32_16x16x32_bf16 (f32 accumulate).
// GEMM waves compute 64x64 tiles: 16 WMMA per 32-deep K-step on 16 b128
// loads (1 load per WMMA).
// ---------------------------------------------------------------------------

#define DEVFN __device__ __forceinline__

typedef __bf16 bf16;
typedef bf16 v16bf __attribute__((ext_vector_type(16)));
typedef float v8f  __attribute__((ext_vector_type(8)));

static constexpr int Bc = 8;
static constexpr int Nn = 1024;     // sequence length
static constexpr int Cc = 1024;     // channels
static constexpr int Hh = 16;       // heads
static constexpr int Pp = 64;       // prefix tokens
static constexpr int Dd = 64;       // head dim
static constexpr int Mm = Pp + Nn;  // 1088 keys
static constexpr int BN = Bc * Nn;  // 8192 rows

union FragU { v16bf v; uint4 q[2]; };

// A-matrix fragment (16x32 bf16, MxK): lane holds row (lane&15); its 16
// elements are the contiguous K-run [16*half .. 16*half+15].
DEVFN v16bf load_a_frag(const bf16* __restrict__ row, int half) {
  FragU f;
  f.q[0] = *(const uint4*)(row + 16 * half);
  f.q[1] = *(const uint4*)(row + 16 * half + 8);
  return f.v;
}

// B-matrix fragment (32x16 bf16, KxN): lane holds column (lane&15); its 16
// elements are the contiguous K-run starting at 16*half.  Caller passes the
// pointer already offset by 16*half.
DEVFN v16bf load_b_frag32(const bf16* __restrict__ p) {
  FragU f;
  f.q[0] = ((const uint4*)p)[0];
  f.q[1] = ((const uint4*)p)[1];
  return f.v;
}

DEVFN v8f wmma_bf16(v16bf a, v16bf b, v8f c) {
  return __builtin_amdgcn_wmma_f32_16x16x32_bf16(
      /*neg_a=*/false, a, /*neg_b=*/false, b,
      /*c_mod=*/(short)0, c, /*reuse_a=*/false, /*reuse_b=*/false);
}

// ---------------------------------------------------------------------------
// Elementwise helpers
// ---------------------------------------------------------------------------
__global__ void f32_to_bf16_kernel(const float* __restrict__ in,
                                   bf16* __restrict__ out, int n) {
  int i = blockIdx.x * 256 + threadIdx.x;
  if (i < n) out[i] = (bf16)in[i];
}

// LayerNorm over C=1024 (f32 in, bf16 out).  One block (256 thr) per row.
__global__ __launch_bounds__(256)
void ln_kernel(const float* __restrict__ x, const float* __restrict__ g,
               const float* __restrict__ be, bf16* __restrict__ out) {
  __shared__ float ssum[8], ssq[8];
  const int row = blockIdx.x;
  const float4 v = ((const float4*)(x + (size_t)row * Cc))[threadIdx.x];
  float s = v.x + v.y + v.z + v.w;
  float q = v.x * v.x + v.y * v.y + v.z * v.z + v.w * v.w;
  for (int off = 16; off; off >>= 1) {
    s += __shfl_xor(s, off, 32);
    q += __shfl_xor(q, off, 32);
  }
  const int wave = threadIdx.x >> 5, lane = threadIdx.x & 31;
  if (lane == 0) { ssum[wave] = s; ssq[wave] = q; }
  __syncthreads();
  s = 0.f; q = 0.f;
#pragma unroll
  for (int w = 0; w < 8; ++w) { s += ssum[w]; q += ssq[w]; }
  const float mu = s * (1.f / Cc);
  const float var = q * (1.f / Cc) - mu * mu;
  const float rstd = rsqrtf(var + 1e-5f);
  const int idx = threadIdx.x * 4;
  const float4 gv = ((const float4*)g)[threadIdx.x];
  const float4 bv = ((const float4*)be)[threadIdx.x];
  bf16* orow = out + (size_t)row * Cc + idx;
  orow[0] = (bf16)((v.x - mu) * rstd * gv.x + bv.x);
  orow[1] = (bf16)((v.y - mu) * rstd * gv.y + bv.y);
  orow[2] = (bf16)((v.z - mu) * rstd * gv.z + bv.z);
  orow[3] = (bf16)((v.w - mu) * rstd * gv.w + bv.w);
}

// prefix pk/pv (f32) + qkv k/v (bf16) -> kbuf[B,H,M,D], vtbuf[B,H,D,M] (bf16)
__global__ void scatter_kv_kernel(const bf16* __restrict__ qkv,
                                  const float* __restrict__ pk,
                                  const float* __restrict__ pv,
                                  bf16* __restrict__ kbuf,
                                  bf16* __restrict__ vtbuf) {
  int i = blockIdx.x * 256 + threadIdx.x;   // flat over [b][h][m][d]
  if (i >= Bc * Hh * Mm * Dd) return;
  const int d = i & 63;
  const int m = (i >> 6) % Mm;
  const int h = (i / (Dd * Mm)) & 15;
  const int b = i / (Dd * Mm * Hh);
  bf16 kv, vv;
  if (m < Pp) {
    const size_t src = ((size_t)(b * Pp + m)) * Cc + h * Dd + d;
    kv = (bf16)pk[src];
    vv = (bf16)pv[src];
  } else {
    const size_t base = ((size_t)(b * Nn + (m - Pp))) * (3 * Cc) + h * Dd + d;
    kv = qkv[base + Cc];        // k slice
    vv = qkv[base + 2 * Cc];    // v slice
  }
  const size_t bh = (size_t)(b * Hh + h);
  kbuf[(bh * Mm + m) * Dd + d] = kv;
  vtbuf[(bh * Dd + d) * Mm + m] = vv;
}

// ---------------------------------------------------------------------------
// WMMA GEMM:  out[M,N] = A[M,K] x W[N,K]^T   (bf16 inputs, f32 accumulate)
// Epilogue: (+bias[col]) -> (QuickGELU) -> (+residual) -> f32 and/or bf16 out.
// Block: 256 thr = 8 waves as 2x4 wave-tiles of 64x64 -> block tile 128x256.
// Per K-step (32): 16 WMMA on 16 b128 loads.
// ---------------------------------------------------------------------------
template <bool BIAS, bool RES, int ACT, bool OF32, bool OBF>
__global__ __launch_bounds__(256)
void gemm_wmma_kernel(const bf16* __restrict__ A, const bf16* __restrict__ W,
                      const float* __restrict__ bias,
                      const float* __restrict__ res,
                      float* __restrict__ outf, bf16* __restrict__ outb,
                      int K, int NcW) {
  const int lane = threadIdx.x & 31;
  const int wave = threadIdx.x >> 5;
  const int half = lane >> 4;
  const int ln16 = lane & 15;
  const int rowbase = blockIdx.y * 128 + (wave >> 2) * 64;
  const int colbase = blockIdx.x * 256 + (wave & 3) * 64;

  v8f acc[4][4];
#pragma unroll
  for (int i = 0; i < 4; ++i)
#pragma unroll
    for (int j = 0; j < 4; ++j) acc[i][j] = v8f{};

  const bf16* ap[4];
  const bf16* bp[4];
#pragma unroll
  for (int i = 0; i < 4; ++i) {
    ap[i] = A + (size_t)(rowbase + 16 * i + ln16) * K;
    bp[i] = W + (size_t)(colbase + 16 * i + ln16) * K + 16 * half;
  }

#pragma unroll 1
  for (int kk = 0; kk < K; kk += 32) {
    __builtin_prefetch(bp[0] + kk + 512, 0, 0);      // global_prefetch_b8
    __builtin_prefetch(ap[0] + kk + 512, 0, 0);
    v16bf a[4], b[4];
#pragma unroll
    for (int i = 0; i < 4; ++i) a[i] = load_a_frag(ap[i] + kk, half);
#pragma unroll
    for (int j = 0; j < 4; ++j) b[j] = load_b_frag32(bp[j] + kk);
#pragma unroll
    for (int i = 0; i < 4; ++i)
#pragma unroll
      for (int j = 0; j < 4; ++j) acc[i][j] = wmma_bf16(a[i], b[j], acc[i][j]);
  }

#pragma unroll
  for (int i = 0; i < 4; ++i) {
#pragma unroll
    for (int j = 0; j < 4; ++j) {
      const v8f c = acc[i][j];
      const int col = colbase + 16 * j + ln16;
      float bv = 0.f;
      if constexpr (BIAS) bv = bias[col];
#pragma unroll
      for (int r = 0; r < 8; ++r) {
        const int row = rowbase + 16 * i + r + 8 * half;
        float v = c[r] + bv;
        if constexpr (ACT == 1) v = v / (1.f + __expf(-1.702f * v)); // QuickGELU
        if constexpr (RES) v += res[(size_t)row * NcW + col];
        if constexpr (OF32) outf[(size_t)row * NcW + col] = v;
        if constexpr (OBF) outb[(size_t)row * NcW + col] = (bf16)v;
      }
    }
  }
}

// ---------------------------------------------------------------------------
// Flash attention.  Block = 8 waves; each wave owns 16 query rows of one
// (b,h); loops over 34 key tiles of 32.  Per tile: 4 WMMA for S = Q K^T,
// online softmax (16-lane shuffle reductions), P staged through a per-wave
// 16x32 bf16 LDS tile, 4 WMMA for O += P V (V pre-transposed [B,H,D,M]).
// Q is read directly from the qkv buffer [B,N,3,H,D] (row stride 3C).
// ---------------------------------------------------------------------------
__global__ __launch_bounds__(256)
void flash_attn_kernel(const bf16* __restrict__ qkvb,
                       const bf16* __restrict__ kbuf,
                       const bf16* __restrict__ vtbuf,
                       bf16* __restrict__ out /* [B,N,C] */) {
  __shared__ __align__(16) bf16 pshare[8][16][32];
  const int lane = threadIdx.x & 31;
  const int wave = threadIdx.x >> 5;
  const int half = lane >> 4;
  const int ln16 = lane & 15;
  const int bh = blockIdx.y;
  const int b = bh >> 4, h = bh & 15;
  const int qstart = blockIdx.x * 128 + wave * 16;

  const bf16* kb = kbuf + (size_t)bh * Mm * Dd;
  const bf16* vt = vtbuf + (size_t)bh * Dd * Mm;

  // Q fragments straight from qkv[B,N,3C] (q slice, head h); kept in regs.
  const bf16* qrow =
      qkvb + (size_t)(b * Nn + qstart + ln16) * (3 * Cc) + h * Dd;
  const v16bf qa0 = load_a_frag(qrow, half);        // d = 0..31
  const v16bf qa1 = load_a_frag(qrow + 32, half);   // d = 32..63

  float mrun[8], lrun[8];
#pragma unroll
  for (int r = 0; r < 8; ++r) { mrun[r] = -3.0e38f; lrun[r] = 0.f; }
  v8f o0{}, o1{}, o2{}, o3{};

  bf16(*pl)[32] = pshare[wave];

  for (int t = 0; t < Mm / 32; ++t) {
    const int k0 = t * 32;
    // ---- S = Q K^T (scaled) ----
    const bf16* krow0 = kb + (size_t)(k0 + ln16) * Dd + 16 * half;
    const bf16* krow1 = kb + (size_t)(k0 + 16 + ln16) * Dd + 16 * half;
    v8f s0{}, s1{};
    s0 = wmma_bf16(qa0, load_b_frag32(krow0), s0);
    s0 = wmma_bf16(qa1, load_b_frag32(krow0 + 32), s0);
    s1 = wmma_bf16(qa0, load_b_frag32(krow1), s1);
    s1 = wmma_bf16(qa1, load_b_frag32(krow1 + 32), s1);

    // ---- online softmax over this 32-key tile ----
    float corr[8];
#pragma unroll
    for (int r = 0; r < 8; ++r) {
      const float x0 = s0[r] * 0.125f;   // D^-0.5
      const float x1 = s1[r] * 0.125f;
      float mx = fmaxf(x0, x1);
      mx = fmaxf(mx, __shfl_xor(mx, 8, 32));
      mx = fmaxf(mx, __shfl_xor(mx, 4, 32));
      mx = fmaxf(mx, __shfl_xor(mx, 2, 32));
      mx = fmaxf(mx, __shfl_xor(mx, 1, 32));
      const float mnew = fmaxf(mrun[r], mx);
      const float p0 = __expf(x0 - mnew);
      const float p1 = __expf(x1 - mnew);
      float rs = p0 + p1;
      rs += __shfl_xor(rs, 8, 32);
      rs += __shfl_xor(rs, 4, 32);
      rs += __shfl_xor(rs, 2, 32);
      rs += __shfl_xor(rs, 1, 32);
      const float cr = __expf(mrun[r] - mnew);
      lrun[r] = lrun[r] * cr + rs;
      mrun[r] = mnew;
      corr[r] = cr;
      const int prow = r + 8 * half;     // C-layout row held by this lane
      pl[prow][ln16] = (bf16)p0;
      pl[prow][ln16 + 16] = (bf16)p1;
    }
    __syncthreads();   // make P visible across lanes (and compiler barrier)

#pragma unroll
    for (int r = 0; r < 8; ++r) {
      o0[r] *= corr[r]; o1[r] *= corr[r]; o2[r] *= corr[r]; o3[r] *= corr[r];
    }

    // ---- O += P V ----  (P re-read in A-layout: two ds_load_b128 per lane)
    FragU pf;
    pf.q[0] = *(const uint4*)(&pl[ln16][16 * half]);
    pf.q[1] = *(const uint4*)(&pl[ln16][16 * half + 8]);
    const v16bf pa = pf.v;
    const size_t ko = (size_t)k0 + 16 * half;
    o0 = wmma_bf16(pa, load_b_frag32(vt + (size_t)(ln16) * Mm + ko), o0);
    o1 = wmma_bf16(pa, load_b_frag32(vt + (size_t)(16 + ln16) * Mm + ko), o1);
    o2 = wmma_bf16(pa, load_b_frag32(vt + (size_t)(32 + ln16) * Mm + ko), o2);
    o3 = wmma_bf16(pa, load_b_frag32(vt + (size_t)(48 + ln16) * Mm + ko), o3);
    __syncthreads();
  }

  // ---- epilogue: normalize and scatter to [B,N,C] bf16 for proj GEMM ----
  float inv[8];
#pragma unroll
  for (int r = 0; r < 8; ++r) inv[r] = 1.f / lrun[r];
  v8f* ofr[4] = {&o0, &o1, &o2, &o3};
#pragma unroll
  for (int g = 0; g < 4; ++g) {
    const v8f og = *ofr[g];
    const int d = 16 * g + ln16;
#pragma unroll
    for (int r = 0; r < 8; ++r) {
      const int q = qstart + r + 8 * half;
      out[((size_t)(b * Nn + q)) * Cc + h * Dd + d] = (bf16)(og[r] * inv[r]);
    }
  }
}

// ---------------------------------------------------------------------------
// Host orchestration
// ---------------------------------------------------------------------------
extern "C" void kernel_launch(void* const* d_in, const int* in_sizes, int n_in,
                              void* d_out, int out_size, void* d_ws,
                              size_t ws_size, hipStream_t stream) {
  const float* x       = (const float*)d_in[0];
  const float* pk      = (const float*)d_in[1];
  const float* pv      = (const float*)d_in[2];
  const float* qkv_w   = (const float*)d_in[3];
  const float* proj_w  = (const float*)d_in[4];
  const float* proj_b  = (const float*)d_in[5];
  const float* ln1_g   = (const float*)d_in[6];
  const float* ln1_b   = (const float*)d_in[7];
  const float* ln2_g   = (const float*)d_in[8];
  const float* ln2_b   = (const float*)d_in[9];
  const float* fc_w    = (const float*)d_in[10];
  const float* fc_b    = (const float*)d_in[11];
  const float* cproj_w = (const float*)d_in[12];
  const float* cproj_b = (const float*)d_in[13];
  float* out = (float*)d_out;

  // Workspace carve-up (256B aligned)
  char* w = (char*)d_ws;
  size_t off = 0;
  auto carve = [&](size_t bytes) {
    char* p = w + off;
    off += (bytes + 255) & ~(size_t)255;
    return p;
  };
  bf16* h1     = (bf16*)carve((size_t)BN * Cc * 2);
  bf16* qkvw   = (bf16*)carve((size_t)3 * Cc * Cc * 2);
  bf16* projw  = (bf16*)carve((size_t)Cc * Cc * 2);
  bf16* fcw    = (bf16*)carve((size_t)4 * Cc * Cc * 2);
  bf16* cprojw = (bf16*)carve((size_t)4 * Cc * Cc * 2);
  bf16* qkvb   = (bf16*)carve((size_t)BN * 3 * Cc * 2);
  bf16* kbuf   = (bf16*)carve((size_t)Bc * Hh * Mm * Dd * 2);
  bf16* vtbuf  = (bf16*)carve((size_t)Bc * Hh * Dd * Mm * 2);
  bf16* attnb  = (bf16*)carve((size_t)BN * Cc * 2);
  float* x1    = (float*)carve((size_t)BN * Cc * 4);
  bf16* h2     = (bf16*)carve((size_t)BN * Cc * 2);
  bf16* gbuf   = (bf16*)carve((size_t)BN * 4 * Cc * 2);
  (void)ws_size; (void)n_in; (void)in_sizes; (void)out_size;

  // 1) weights -> bf16
  f32_to_bf16_kernel<<<(3 * Cc * Cc) / 256, 256, 0, stream>>>(qkv_w, qkvw, 3 * Cc * Cc);
  f32_to_bf16_kernel<<<(Cc * Cc) / 256, 256, 0, stream>>>(proj_w, projw, Cc * Cc);
  f32_to_bf16_kernel<<<(4 * Cc * Cc) / 256, 256, 0, stream>>>(fc_w, fcw, 4 * Cc * Cc);
  f32_to_bf16_kernel<<<(4 * Cc * Cc) / 256, 256, 0, stream>>>(cproj_w, cprojw, 4 * Cc * Cc);

  // 2) LN1
  ln_kernel<<<BN, 256, 0, stream>>>(x, ln1_g, ln1_b, h1);

  // 3) qkv = h1 @ qkv_w^T  (bf16 out)
  gemm_wmma_kernel<false, false, 0, false, true>
      <<<dim3((3 * Cc) / 256, BN / 128), 256, 0, stream>>>(
          h1, qkvw, nullptr, nullptr, nullptr, qkvb, Cc, 3 * Cc);

  // 4) scatter (prefix+)kv, V transposed; Q is consumed in-place from qkvb
  scatter_kv_kernel<<<(Bc * Hh * Mm * Dd + 255) / 256, 256, 0, stream>>>(
      qkvb, pk, pv, kbuf, vtbuf);

  // 5) flash attention -> [B,N,C] bf16
  flash_attn_kernel<<<dim3(Nn / 128, Bc * Hh), 256, 0, stream>>>(
      qkvb, kbuf, vtbuf, attnb);

  // 6) x1 = x + attn @ proj_w^T + proj_b   (f32 out)
  gemm_wmma_kernel<true, true, 0, true, false>
      <<<dim3(Cc / 256, BN / 128), 256, 0, stream>>>(
          attnb, projw, proj_b, x, x1, nullptr, Cc, Cc);

  // 7) LN2
  ln_kernel<<<BN, 256, 0, stream>>>(x1, ln2_g, ln2_b, h2);

  // 8) g = QuickGELU(h2 @ fc_w^T + fc_b)   (bf16 out)
  gemm_wmma_kernel<true, false, 1, false, true>
      <<<dim3((4 * Cc) / 256, BN / 128), 256, 0, stream>>>(
          h2, fcw, fc_b, nullptr, nullptr, gbuf, Cc, 4 * Cc);

  // 9) out = x1 + g @ cproj_w^T + cproj_b  (f32 out)
  gemm_wmma_kernel<true, true, 0, true, false>
      <<<dim3(Cc / 256, BN / 128), 256, 0, stream>>>(
          gbuf, cprojw, cproj_b, x1, out, nullptr, 4 * Cc, Cc);
}
